// MultiRelGraphConv_7404523618515
// MI455X (gfx1250) — compile-verified, matrix operand order checked by reading, not run on hardware
//
#include <hip/hip_runtime.h>
#include <math.h>

#define N_NODESC 50000
#define N_EDGESC 500000
#define DIM 128
#define K3D 384
#define NRELC 200
#define WPR 1568 /* ceil(50000/32)=1563, padded to 32B multiple */
#define RRELU_SLOPE 0.22916666666666666f

typedef __attribute__((ext_vector_type(16))) __bf16 v16bf;
typedef __attribute__((ext_vector_type(8)))  float  v8f;

union AFrag { v16bf v; unsigned short u[16]; };

__device__ __forceinline__ unsigned short f2bf(float x) {
  unsigned int u = __float_as_uint(x);
  return (unsigned short)((u + 0x7FFFu + ((u >> 16) & 1u)) >> 16);
}
__device__ __forceinline__ float sigm(float x) { return 1.0f / (1.0f + __expf(-x)); }

// K offset inside a 16x32 bf16 A/B fragment for (lane, element) per CDNA5 ISA layout.
__device__ __forceinline__ int wmma_k(int lane, int e) {
  int v = e >> 1, h = e & 1;
  return ((v & 4) << 2) + ((lane >> 4) << 3) + ((v & 3) << 1) + h;
}

// ---------------- utility kernels ----------------
__global__ void k_zero_f(float* p, long n) {
  long i = (long)blockIdx.x * blockDim.x + threadIdx.x;
  if (i < n) p[i] = 0.0f;
}
__global__ void k_zero_u(unsigned* p, long n) {
  long i = (long)blockIdx.x * blockDim.x + threadIdx.x;
  if (i < n) p[i] = 0u;
}

// relation membership bitmask: mask[r][node]
__global__ void k_mask(const int* __restrict__ src, const int* __restrict__ dst,
                       const int* __restrict__ et, unsigned* __restrict__ mask) {
  int e = blockIdx.x * blockDim.x + threadIdx.x;
  if (e >= N_EDGESC) return;
  int r = et[e], s = src[e], d = dst[e];
  atomicOr(mask + (size_t)r * WPR + (s >> 5), 1u << (s & 31));
  atomicOr(mask + (size_t)r * WPR + (d >> 5), 1u << (d & 31));
}

__global__ void k_cnt(const unsigned* __restrict__ mask, float* __restrict__ cnt) {
  int r = blockIdx.x;
  int acc = 0;
  for (int w = threadIdx.x; w < WPR; w += blockDim.x)
    acc += __popc(mask[(size_t)r * WPR + w]);
  atomicAdd(&cnt[r], (float)acc);
}

// block(128 threads) = (rel = blockIdx.y, 256-node chunk = blockIdx.x); thread = feature dim
__global__ void k_relsum(const unsigned* __restrict__ mask, const float* __restrict__ nf,
                         float* __restrict__ rel_sum) {
  int r = blockIdx.y, t = threadIdx.x, base = blockIdx.x * 256;
  float s = 0.0f;
  for (int i = 0; i < 256; i++) {
    int n = base + i;
    if (n >= N_NODESC) break;
    unsigned w = mask[(size_t)r * WPR + (n >> 5)];
    if (w & (1u << (n & 31))) s += nf[(size_t)n * DIM + t];
  }
  atomicAdd(&rel_sum[r * DIM + t], s);
}

__global__ void k_relfinal(const float* __restrict__ rel_sum, const float* __restrict__ cnt,
                           float* __restrict__ rel_emb) {
  int i = blockIdx.x * blockDim.x + threadIdx.x;
  if (i >= NRELC * DIM) return;
  float c = cnt[i / DIM];
  rel_emb[i] = (c > 0.0f) ? rel_sum[i] / c : 0.0f;
}

__global__ void k_deg(const int* __restrict__ dst, float* __restrict__ deg) {
  int e = blockIdx.x * blockDim.x + threadIdx.x;
  if (e < N_EDGESC) atomicAdd(&deg[dst[e]], 1.0f);
}

// pack weight W (N rows x K cols, row-major; W[n*K+k]) into per-wave bf16 B-fragments:
// frag f = kt*(N/16)+nt, layout out[((f*32 + lane)*16) + e]
__global__ void k_packW(const float* __restrict__ W, unsigned short* __restrict__ out,
                        int K, int N) {
  int p = blockIdx.x * blockDim.x + threadIdx.x;
  if (p >= K * N) return;
  int f = p >> 9, rem = p & 511, lane = rem >> 4, e = rem & 15;
  int ntiles = N >> 4;
  int kt = f / ntiles, nt = f - kt * ntiles;
  int n = nt * 16 + (lane & 15);
  int k = kt * 32 + wmma_k(lane, e);
  out[p] = f2bf(W[(size_t)n * K + k]);
}

// ---------------- WMMA GEMM kernels ----------------
__device__ __forceinline__ v8f gemm_acc(const AFrag* a, int nk,
                                        const unsigned short* __restrict__ packB,
                                        int ntile, int NT, int lane) {
  v8f acc = {0.f, 0.f, 0.f, 0.f, 0.f, 0.f, 0.f, 0.f};
#pragma unroll
  for (int kt = 0; kt < 4; kt++) {
    if (kt >= nk) break;
    const v16bf b = *(const v16bf*)(packB + ((size_t)(kt * NT + ntile) * 32 + lane) * 16);
    acc = __builtin_amdgcn_wmma_f32_16x16x32_bf16(false, a[kt].v, false, b, (short)0, acc,
                                                  false, false);
  }
  return acc;
}

// msg = [re|hs|ef] @ W1^T, scatter-add into node_msg[dst] (bias b1 folded into node kernel)
__global__ __launch_bounds__(128) void k_edge_msg(
    const float* __restrict__ nf, const float* __restrict__ ef,
    const float* __restrict__ rel_emb, const int* __restrict__ src,
    const int* __restrict__ dst, const int* __restrict__ et,
    const unsigned short* __restrict__ packW1, float* __restrict__ node_msg) {
  __shared__ unsigned short xa[64][K3D];
  __shared__ int sdst[64];
  int tid = threadIdx.x;
  long eBase = (long)blockIdx.x * 64;
  if (tid == 0) __builtin_prefetch(ef + (size_t)(eBase + 64) * DIM, 0, 1);
  for (int idx = tid; idx < 64 * K3D; idx += 128) {
    int row = idx / K3D, col = idx - row * K3D;
    long e = eBase + row; if (e >= N_EDGESC) e = N_EDGESC - 1;
    float val;
    if (col < DIM)            val = rel_emb[(size_t)et[e] * DIM + col];
    else if (col < 2 * DIM)   val = nf[(size_t)src[e] * DIM + (col - DIM)];
    else                      val = ef[(size_t)e * DIM + (col - 2 * DIM)];
    xa[row][col] = f2bf(val);
  }
  for (int idx = tid; idx < 64; idx += 128) {
    long e = eBase + idx;
    sdst[idx] = (e < N_EDGESC) ? dst[e] : 0;
  }
  __syncthreads();
  int wave = tid >> 5, lane = tid & 31;
  long tile = (long)blockIdx.x * 4 + wave;
  if (tile * 16 >= N_EDGESC) return;  // wave-uniform
  AFrag a[12];
  int m = lane & 15;
#pragma unroll
  for (int kt = 0; kt < 12; kt++)
#pragma unroll
    for (int e2 = 0; e2 < 16; e2++)
      a[kt].u[e2] = xa[wave * 16 + m][kt * 32 + wmma_k(lane, e2)];
#pragma unroll
  for (int nt = 0; nt < 8; nt++) {
    v8f acc = {0.f, 0.f, 0.f, 0.f, 0.f, 0.f, 0.f, 0.f};
#pragma unroll
    for (int kt = 0; kt < 12; kt++) {
      const v16bf b = *(const v16bf*)(packW1 + ((size_t)(kt * 8 + nt) * 32 + lane) * 16);
      acc = __builtin_amdgcn_wmma_f32_16x16x32_bf16(false, a[kt].v, false, b, (short)0, acc,
                                                    false, false);
    }
    int col = nt * 16 + (lane & 15);
    int mb = (lane >> 4) << 3;
#pragma unroll
    for (int v = 0; v < 8; v++) {
      long e = tile * 16 + mb + v;
      if (e < N_EDGESC)
        atomicAdd(&node_msg[(size_t)sdst[wave * 16 + mb + v] * DIM + col], acc[v]);
    }
  }
}

// edge_msg = [re|hs|hd] @ W2^T + b2, stored bf16 for the GRU input GEMM
__global__ __launch_bounds__(128) void k_edge_emsg(
    const float* __restrict__ nf, const float* __restrict__ rel_emb,
    const int* __restrict__ src, const int* __restrict__ dst, const int* __restrict__ et,
    const unsigned short* __restrict__ packW2, const float* __restrict__ b2,
    unsigned short* __restrict__ emsg) {
  __shared__ unsigned short xa[64][K3D];
  int tid = threadIdx.x;
  long eBase = (long)blockIdx.x * 64;
  for (int idx = tid; idx < 64 * K3D; idx += 128) {
    int row = idx / K3D, col = idx - row * K3D;
    long e = eBase + row; if (e >= N_EDGESC) e = N_EDGESC - 1;
    float val;
    if (col < DIM)            val = rel_emb[(size_t)et[e] * DIM + col];
    else if (col < 2 * DIM)   val = nf[(size_t)src[e] * DIM + (col - DIM)];
    else                      val = nf[(size_t)dst[e] * DIM + (col - 2 * DIM)];
    xa[row][col] = f2bf(val);
  }
  __syncthreads();
  int wave = tid >> 5, lane = tid & 31;
  long tile = (long)blockIdx.x * 4 + wave;
  if (tile * 16 >= N_EDGESC) return;
  AFrag a[12];
  int m = lane & 15;
#pragma unroll
  for (int kt = 0; kt < 12; kt++)
#pragma unroll
    for (int e2 = 0; e2 < 16; e2++)
      a[kt].u[e2] = xa[wave * 16 + m][kt * 32 + wmma_k(lane, e2)];
#pragma unroll
  for (int nt = 0; nt < 8; nt++) {
    v8f acc = {0.f, 0.f, 0.f, 0.f, 0.f, 0.f, 0.f, 0.f};
#pragma unroll
    for (int kt = 0; kt < 12; kt++) {
      const v16bf b = *(const v16bf*)(packW2 + ((size_t)(kt * 8 + nt) * 32 + lane) * 16);
      acc = __builtin_amdgcn_wmma_f32_16x16x32_bf16(false, a[kt].v, false, b, (short)0, acc,
                                                    false, false);
    }
    int col = nt * 16 + (lane & 15);
    float bc = b2[col];
    int mb = (lane >> 4) << 3;
#pragma unroll
    for (int v = 0; v < 8; v++) {
      long e = tile * 16 + mb + v;
      if (e < N_EDGESC) emsg[(size_t)e * DIM + col] = f2bf(acc[v] + bc);
    }
  }
}

// GRU: gi = emsg @ Wih^T + bih ; gh = ef @ Whh^T + bhh ; gates; new_e
__global__ __launch_bounds__(128) void k_gru(
    const unsigned short* __restrict__ emsg, const float* __restrict__ ef,
    const unsigned short* __restrict__ packWih, const unsigned short* __restrict__ packWhh,
    const float* __restrict__ bih, const float* __restrict__ bhh,
    float* __restrict__ ef_out) {
  __shared__ unsigned short xm[64][DIM];
  __shared__ unsigned short xh[64][DIM];
  int tid = threadIdx.x;
  long eBase = (long)blockIdx.x * 64;
  for (int idx = tid; idx < 64 * DIM; idx += 128) {
    int row = idx >> 7, col = idx & 127;
    long e = eBase + row; if (e >= N_EDGESC) e = N_EDGESC - 1;
    xm[row][col] = emsg[(size_t)e * DIM + col];
    xh[row][col] = f2bf(ef[(size_t)e * DIM + col]);
  }
  __syncthreads();
  int wave = tid >> 5, lane = tid & 31;
  long tile = (long)blockIdx.x * 4 + wave;
  if (tile * 16 >= N_EDGESC) return;
  AFrag am[4], ah[4];
  int m = lane & 15;
#pragma unroll
  for (int kt = 0; kt < 4; kt++)
#pragma unroll
    for (int e2 = 0; e2 < 16; e2++) {
      int k = kt * 32 + wmma_k(lane, e2);
      am[kt].u[e2] = xm[wave * 16 + m][k];
      ah[kt].u[e2] = xh[wave * 16 + m][k];
    }
  int nc = lane & 15, mb = (lane >> 4) << 3;
#pragma unroll
  for (int nt = 0; nt < 8; nt++) {
    v8f ir = gemm_acc(am, 4, packWih, nt, 24, lane);
    v8f hr = gemm_acc(ah, 4, packWhh, nt, 24, lane);
    v8f iz = gemm_acc(am, 4, packWih, 8 + nt, 24, lane);
    v8f hz = gemm_acc(ah, 4, packWhh, 8 + nt, 24, lane);
    v8f in_ = gemm_acc(am, 4, packWih, 16 + nt, 24, lane);
    v8f hn = gemm_acc(ah, 4, packWhh, 16 + nt, 24, lane);
    int col = nt * 16 + nc;
    float bir = bih[col], bhr = bhh[col];
    float biz = bih[128 + col], bhz = bhh[128 + col];
    float bin = bih[256 + col], bhn = bhh[256 + col];
#pragma unroll
    for (int v = 0; v < 8; v++) {
      long e = tile * 16 + mb + v;
      if (e >= N_EDGESC) continue;
      float r = sigm(ir[v] + bir + hr[v] + bhr);
      float z = sigm(iz[v] + biz + hz[v] + bhz);
      float n = tanhf(in_[v] + bin + r * (hn[v] + bhn));
      float h = ef[(size_t)e * DIM + col];
      ef_out[(size_t)e * DIM + col] = (1.0f - z) * n + z * h;
    }
  }
}

// new_n = leaky(node_msg/deg + b1 [if deg>0] + nf@W3^T + b3)
__global__ __launch_bounds__(128) void k_node(
    const float* __restrict__ nf, const float* __restrict__ node_msg,
    const float* __restrict__ deg, const unsigned short* __restrict__ packW3,
    const float* __restrict__ b1, const float* __restrict__ b3, float* __restrict__ out) {
  __shared__ unsigned short xn[64][DIM];
  int tid = threadIdx.x;
  long nBase = (long)blockIdx.x * 64;
  for (int idx = tid; idx < 64 * DIM; idx += 128) {
    int row = idx >> 7, col = idx & 127;
    long n = nBase + row; if (n >= N_NODESC) n = N_NODESC - 1;
    xn[row][col] = f2bf(nf[(size_t)n * DIM + col]);
  }
  __syncthreads();
  int wave = tid >> 5, lane = tid & 31;
  long tile = (long)blockIdx.x * 4 + wave;
  if (tile * 16 >= N_NODESC) return;
  AFrag a[4];
  int m = lane & 15;
#pragma unroll
  for (int kt = 0; kt < 4; kt++)
#pragma unroll
    for (int e2 = 0; e2 < 16; e2++)
      a[kt].u[e2] = xn[wave * 16 + m][kt * 32 + wmma_k(lane, e2)];
  int mb = (lane >> 4) << 3;
#pragma unroll
  for (int nt = 0; nt < 8; nt++) {
    v8f acc = gemm_acc(a, 4, packW3, nt, 8, lane);
    int col = nt * 16 + (lane & 15);
    float b3c = b3[col], b1c = b1[col];
#pragma unroll
    for (int v = 0; v < 8; v++) {
      long n = tile * 16 + mb + v;
      if (n >= N_NODESC) continue;
      float dg = deg[n];
      float nm = (dg > 0.0f) ? node_msg[(size_t)n * DIM + col] / dg + b1c : 0.0f;
      float pre = acc[v] + nm + b3c;
      out[(size_t)n * DIM + col] = (pre >= 0.0f) ? pre : RRELU_SLOPE * pre;
    }
  }
}

// ---------------- host launch ----------------
extern "C" void kernel_launch(void* const* d_in, const int* in_sizes, int n_in,
                              void* d_out, int out_size, void* d_ws, size_t ws_size,
                              hipStream_t stream) {
  const float* nf0 = (const float*)d_in[0];
  const float* ef0 = (const float*)d_in[1];
  const int* src = (const int*)d_in[2];
  const int* dst = (const int*)d_in[3];
  const int* et  = (const int*)d_in[4];
  const float* W1 = (const float*)d_in[6];
  const float* b1 = (const float*)d_in[7];
  const float* W2 = (const float*)d_in[8];
  const float* b2 = (const float*)d_in[9];
  const float* W3 = (const float*)d_in[10];
  const float* b3 = (const float*)d_in[11];
  const float* Wih = (const float*)d_in[12];
  const float* Whh = (const float*)d_in[13];
  const float* bih = (const float*)d_in[14];
  const float* bhh = (const float*)d_in[15];

  char* ws = (char*)d_ws;
  size_t off = 0;
  auto take = [&](size_t bytes) {
    void* p = ws + off;
    off += (bytes + 255) & ~(size_t)255;
    return p;
  };
  unsigned* mask   = (unsigned*)take((size_t)NRELC * WPR * 4);
  float* cnt       = (float*)take(NRELC * 4);
  float* rel_sum   = (float*)take((size_t)NRELC * DIM * 4);
  float* rel_emb   = (float*)take((size_t)NRELC * DIM * 4);
  float* deg       = (float*)take((size_t)N_NODESC * 4);
  float* node_msg  = (float*)take((size_t)N_NODESC * DIM * 4);
  float* nf1       = (float*)take((size_t)N_NODESC * DIM * 4);
  float* ef1       = (float*)take((size_t)N_EDGESC * DIM * 4);
  unsigned short* emsg = (unsigned short*)take((size_t)N_EDGESC * DIM * 2);
  unsigned short* pW1_0 = (unsigned short*)take((size_t)K3D * DIM * 2);
  unsigned short* pW1_1 = (unsigned short*)take((size_t)K3D * DIM * 2);
  unsigned short* pW2_0 = (unsigned short*)take((size_t)K3D * DIM * 2);
  unsigned short* pW3_0 = (unsigned short*)take((size_t)DIM * DIM * 2);
  unsigned short* pW3_1 = (unsigned short*)take((size_t)DIM * DIM * 2);
  unsigned short* pWih  = (unsigned short*)take((size_t)DIM * K3D * 2);
  unsigned short* pWhh  = (unsigned short*)take((size_t)DIM * K3D * 2);

  const int ZT = 256;
  auto zgrid = [](long n) { return (unsigned)((n + 255) / 256); };
  const long maskN = (long)NRELC * WPR;
  const long nmN = (long)N_NODESC * DIM;
  const int eBlocks = (N_EDGESC + 255) / 256;
  const int edgeTiles = N_EDGESC / 16;          // 31250
  const int edgeGemmBlocks = (edgeTiles + 3) / 4;  // 7813
  const int nodeTiles = (N_NODESC + 15) / 16;   // 3125
  const int nodeGemmBlocks = (nodeTiles + 3) / 4;  // 782

  // ---- one-time setup ----
  k_zero_u<<<zgrid(maskN), ZT, 0, stream>>>(mask, maskN);
  k_zero_f<<<zgrid(NRELC), ZT, 0, stream>>>(cnt, NRELC);
  k_zero_f<<<zgrid(N_NODESC), ZT, 0, stream>>>(deg, N_NODESC);
  k_mask<<<eBlocks, 256, 0, stream>>>(src, dst, et, mask);
  k_deg<<<eBlocks, 256, 0, stream>>>(dst, deg);
  k_cnt<<<NRELC, 128, 0, stream>>>(mask, cnt);
  k_packW<<<(K3D * DIM + 255) / 256, 256, 0, stream>>>(W1, pW1_0, K3D, DIM);
  k_packW<<<(K3D * DIM + 255) / 256, 256, 0, stream>>>(W1 + (size_t)DIM * K3D, pW1_1, K3D, DIM);
  k_packW<<<(K3D * DIM + 255) / 256, 256, 0, stream>>>(W2, pW2_0, K3D, DIM);
  k_packW<<<(DIM * DIM + 255) / 256, 256, 0, stream>>>(W3, pW3_0, DIM, DIM);
  k_packW<<<(DIM * DIM + 255) / 256, 256, 0, stream>>>(W3 + (size_t)DIM * DIM, pW3_1, DIM, DIM);
  k_packW<<<(DIM * K3D + 255) / 256, 256, 0, stream>>>(Wih, pWih, DIM, K3D);
  k_packW<<<(DIM * K3D + 255) / 256, 256, 0, stream>>>(Whh, pWhh, DIM, K3D);

  dim3 rsGrid((N_NODESC + 255) / 256, NRELC);

  // ---- layer 0 ----
  k_zero_f<<<zgrid((long)NRELC * DIM), ZT, 0, stream>>>(rel_sum, (long)NRELC * DIM);
  k_zero_f<<<zgrid(nmN), ZT, 0, stream>>>(node_msg, nmN);
  k_relsum<<<rsGrid, 128, 0, stream>>>(mask, nf0, rel_sum);
  k_relfinal<<<(NRELC * DIM + 255) / 256, 256, 0, stream>>>(rel_sum, cnt, rel_emb);
  k_edge_msg<<<edgeGemmBlocks, 128, 0, stream>>>(nf0, ef0, rel_emb, src, dst, et, pW1_0,
                                                 node_msg);
  k_edge_emsg<<<edgeGemmBlocks, 128, 0, stream>>>(nf0, rel_emb, src, dst, et, pW2_0, b2, emsg);
  k_gru<<<edgeGemmBlocks, 128, 0, stream>>>(emsg, ef0, pWih, pWhh, bih, bhh, ef1);
  k_node<<<nodeGemmBlocks, 128, 0, stream>>>(nf0, node_msg, deg, pW3_0, b1, b3, nf1);

  // ---- layer 1 (GRU output unused -> skipped) ----
  k_zero_f<<<zgrid((long)NRELC * DIM), ZT, 0, stream>>>(rel_sum, (long)NRELC * DIM);
  k_zero_f<<<zgrid(nmN), ZT, 0, stream>>>(node_msg, nmN);
  k_relsum<<<rsGrid, 128, 0, stream>>>(mask, nf1, rel_sum);
  k_relfinal<<<(NRELC * DIM + 255) / 256, 256, 0, stream>>>(rel_sum, cnt, rel_emb);
  k_edge_msg<<<edgeGemmBlocks, 128, 0, stream>>>(nf1, ef1, rel_emb, src, dst, et, pW1_1,
                                                 node_msg);
  k_node<<<nodeGemmBlocks, 128, 0, stream>>>(nf1, node_msg, deg, pW3_1, b1 + DIM, b3 + DIM,
                                             (float*)d_out);
}
